// HNN_89464168776129
// MI455X (gfx1250) — matrix operand order repaired and checked by reading, not compiled, not run on hardware
//
#include <hip/hip_runtime.h>
#include <hip/hip_bf16.h>

// ---------------------------------------------------------------------------
// Fused Hamiltonian-MLP gradient for MI455X (gfx1250, wave32, WMMA bf16).
// Each wave owns a 16-row batch tile. All activations live in 3 wave-private
// LDS regions (48 KB/wave); 4 waves/WG = one wave per SIMD32 of the WGP.
// Six GEMMs (3 fwd + 3 bwd) run on v_wmma_f32_16x16x32_bf16, f32 accumulate,
// with all 16 A-fragments hoisted to VGPRs per layer and two independent
// accumulator chains to overlap WMMA latency.
// ---------------------------------------------------------------------------

typedef __attribute__((ext_vector_type(16))) __bf16        v16bf;
typedef __attribute__((ext_vector_type(8)))  float         v8f;
typedef __attribute__((ext_vector_type(8)))  unsigned int  v8u;

#define BATCH 65536
#define DIN   64
#define HID   512
#define WPB   4                      // waves per workgroup (1 per SIMD32)
#define TILE_ELEMS (16 * HID)        // one 16-row activation tile (bf16 elems)

__device__ __forceinline__ unsigned short f2bf(float f) {
  unsigned int u = __builtin_bit_cast(unsigned int, f);
  u += 0x7FFFu + ((u >> 16) & 1u);              // round-to-nearest-even
  return (unsigned short)(u >> 16);
}
__device__ __forceinline__ float bf2f(unsigned short h) {
  unsigned int u = ((unsigned int)h) << 16;
  return __builtin_bit_cast(float, u);
}

// Branchless fast tanh: 1 - 2/(1+exp(2x)).  v_exp_f32 + v_rcp_f32, no
// EXEC divergence (large |x| saturates to +-1 through inf/zero naturally).
__device__ __forceinline__ float fast_tanh(float x) {
  float e = __expf(2.0f * x);
  return 1.0f - 2.0f * __builtin_amdgcn_rcpf(e + 1.0f);
}

union FragU { v8u u; v16bf b; };

// A fragment: 16x32 bf16 tile from LDS (row stride `stride` elements).
// Contiguous k order used here (same as load_b): lane m=L&15, hi=L>>4; VGPR v holds K pair at
// k = kt*32 + hi*16 + v*2.
__device__ __forceinline__ v16bf load_a(const unsigned short* src, int stride,
                                        int kt, int lane) {
  int m  = lane & 15;
  int hi = lane >> 4;
  const unsigned int* p = (const unsigned int*)(src + m * stride);
  FragU f;
#pragma unroll
  for (int v = 0; v < 8; ++v) {
    int k = (kt << 5) + (hi << 4) + (v << 1);
    f.u[v] = p[k >> 1];
  }
  return f.b;
}

// Hoist the whole 16 x (KT*32) A tile into VGPRs (KT fragments).
template <int KT>
__device__ __forceinline__ void load_a_all(const unsigned short* src, int stride,
                                           int lane, v16bf* af) {
#pragma unroll
  for (int kt = 0; kt < KT; ++kt) af[kt] = load_a(src, stride, kt, lane);
}

// B fragment: 32x16 bf16 tile; W stored as [N][K] bf16 (stride K).
// Lane n=L&15 holds K = kt*32 + hi*16 .. +15 -> one contiguous 32B load.
__device__ __forceinline__ v16bf load_b(const unsigned short* W, int K,
                                        int nt, int kt, int lane) {
  int n  = (nt << 4) + (lane & 15);
  int k0 = (kt << 5) + ((lane >> 4) << 4);
  return *(const v16bf*)(W + n * K + k0);
}

// One 16x16 output tile: A from hoisted fragments, B streamed from global.
// Two independent accumulator chains (even/odd kt) overlap WMMA latency.
template <int KT>
__device__ __forceinline__ v8f gemm_tile(const v16bf* af, const unsigned short* W,
                                         int K, int nt, int lane) {
  v8f acc0 = {}, acc1 = {};
#pragma unroll
  for (int kt = 0; kt < KT; kt += 2) {
    v16bf b0 = load_b(W, K, nt, kt, lane);
    acc0 = __builtin_amdgcn_wmma_f32_16x16x32_bf16(false, af[kt], false, b0,
                                                   (short)0, acc0, false, false);
    if (kt + 1 < KT) {
      v16bf b1 = load_b(W, K, nt, kt + 1, lane);
      acc1 = __builtin_amdgcn_wmma_f32_16x16x32_bf16(false, af[kt + 1], false, b1,
                                                     (short)0, acc1, false, false);
    }
  }
  return acc0 + acc1;
}

// Convert one f32 weight matrix (K x N row-major) to bf16 in two layouts:
//   Wf[n*K + k] = W[k][n]   (B-layout for the forward GEMM)
//   Wr[k*N + n] = W[k][n]   (row-major bf16 == B-layout of W^T for backward)
__global__ void prep_weights(const float* __restrict__ W,
                             unsigned short* __restrict__ Wf,
                             unsigned short* __restrict__ Wr, int K, int N) {
  int idx = blockIdx.x * blockDim.x + threadIdx.x;
  if (idx >= K * N) return;
  int k = idx / N, n = idx - k * N;
  unsigned short b = f2bf(W[idx]);
  Wr[idx]       = b;
  Wf[n * K + k] = b;
}

__global__ void __launch_bounds__(WPB * 32, 1) hnn_fused(
    const float* __restrict__ x,
    const float* __restrict__ b0v, const float* __restrict__ b1v,
    const float* __restrict__ b2v, const float* __restrict__ w3v,
    const unsigned short* __restrict__ W0f, const unsigned short* __restrict__ W0r,
    const unsigned short* __restrict__ W1f, const unsigned short* __restrict__ W1r,
    const unsigned short* __restrict__ W2f, const unsigned short* __restrict__ W2r,
    float* __restrict__ out) {
  extern __shared__ unsigned short lds[];
  const int lane = threadIdx.x & 31;
  const int wave = threadIdx.x >> 5;
  // Three wave-private 16x512 bf16 regions:
  //   r0: h0 -> g0 (in place)   r1: h1 -> g1 (in place)   r2: x-stage -> g2
  unsigned short* r0 = lds + wave * (3 * TILE_ELEMS);
  unsigned short* r1 = r0 + TILE_ELEMS;
  unsigned short* r2 = r1 + TILE_ELEMS;
  const int row0 = (blockIdx.x * WPB + wave) * 16;

  const int nc = lane & 15;   // C/D column-in-tile for this lane
  const int hi = lane >> 4;   // C/D row offset (+8) for upper half-wave

  v16bf af[HID / 32];         // hoisted A fragments (128 VGPRs when KT=16)

  // Stage x tile (16 x 64) into r2 as bf16 (row stride DIN).
  for (int idx = lane; idx < 16 * DIN; idx += 32) {
    int m = idx >> 6, k = idx & (DIN - 1);
    r2[m * DIN + k] = f2bf(x[(row0 + m) * DIN + k]);
  }

  // ---- forward layer 0 (K = 64): h0 = tanh(x@W0+b0) -> r0 ----
  load_a_all<DIN / 32>(r2, DIN, lane, af);
  for (int nt = 0; nt < HID / 16; ++nt) {
    float bias = b0v[nt * 16 + nc];
    v8f acc = gemm_tile<DIN / 32>(af, W0f, DIN, nt, lane);
#pragma unroll
    for (int r = 0; r < 8; ++r) {
      int m = r + hi * 8;
      r0[m * HID + nt * 16 + nc] = f2bf(fast_tanh(acc[r] + bias));
    }
  }
  // ---- forward layer 1: h1 = tanh(h0@W1+b1) -> r1 ----
  load_a_all<HID / 32>(r0, HID, lane, af);
  for (int nt = 0; nt < HID / 16; ++nt) {
    float bias = b1v[nt * 16 + nc];
    v8f acc = gemm_tile<HID / 32>(af, W1f, HID, nt, lane);
#pragma unroll
    for (int r = 0; r < 8; ++r) {
      int m = r + hi * 8;
      r1[m * HID + nt * 16 + nc] = f2bf(fast_tanh(acc[r] + bias));
    }
  }
  // ---- forward layer 2 fused with g2: g2 = (1-tanh^2(pre)) * W3 -> r2 ----
  load_a_all<HID / 32>(r1, HID, lane, af);
  for (int nt = 0; nt < HID / 16; ++nt) {
    float bias = b2v[nt * 16 + nc];
    float w3   = w3v[nt * 16 + nc];
    v8f acc = gemm_tile<HID / 32>(af, W2f, HID, nt, lane);
#pragma unroll
    for (int r = 0; r < 8; ++r) {
      int m = r + hi * 8;
      float t = fast_tanh(acc[r] + bias);
      r2[m * HID + nt * 16 + nc] = f2bf((1.0f - t * t) * w3);
    }
  }
  // ---- backward layer 2: g1 = (1-h1^2) o (g2@W2^T), in place over r1 ----
  load_a_all<HID / 32>(r2, HID, lane, af);
  for (int nt = 0; nt < HID / 16; ++nt) {
    v8f acc = gemm_tile<HID / 32>(af, W2r, HID, nt, lane);
#pragma unroll
    for (int r = 0; r < 8; ++r) {
      int m = r + hi * 8;
      float h = bf2f(r1[m * HID + nt * 16 + nc]);
      r1[m * HID + nt * 16 + nc] = f2bf((1.0f - h * h) * acc[r]);
    }
  }
  // ---- backward layer 1: g0 = (1-h0^2) o (g1@W1^T), in place over r0 ----
  load_a_all<HID / 32>(r1, HID, lane, af);
  for (int nt = 0; nt < HID / 16; ++nt) {
    v8f acc = gemm_tile<HID / 32>(af, W1r, HID, nt, lane);
#pragma unroll
    for (int r = 0; r < 8; ++r) {
      int m = r + hi * 8;
      float h = bf2f(r0[m * HID + nt * 16 + nc]);
      r0[m * HID + nt * 16 + nc] = f2bf((1.0f - h * h) * acc[r]);
    }
  }
  // ---- final: gradH = g0 @ W0^T, symplectic swap/negate, write f32 out ----
  load_a_all<HID / 32>(r0, HID, lane, af);
  for (int nt = 0; nt < DIN / 16; ++nt) {
    v8f acc = gemm_tile<HID / 32>(af, W0r, HID, nt, lane);
    int col  = nt * 16 + nc;
    int ocol = (col + 32) & 63;   // out[:,:32]=grad[:,32:], out[:,32:]=-grad[:,:32]
    float sgn = (col < 32) ? -1.0f : 1.0f;
#pragma unroll
    for (int r = 0; r < 8; ++r) {
      int m = r + hi * 8;
      out[(row0 + m) * DIN + ocol] = sgn * acc[r];
    }
  }
}

extern "C" void kernel_launch(void* const* d_in, const int* in_sizes, int n_in,
                              void* d_out, int out_size, void* d_ws, size_t ws_size,
                              hipStream_t stream) {
  (void)in_sizes; (void)n_in; (void)out_size; (void)ws_size;
  // setup_inputs order: t, x, W0, b0, W1, b1, W2, b2, W3, b3
  const float* x  = (const float*)d_in[1];
  const float* W0 = (const float*)d_in[2];
  const float* b0 = (const float*)d_in[3];
  const float* W1 = (const float*)d_in[4];
  const float* b1 = (const float*)d_in[5];
  const float* W2 = (const float*)d_in[6];
  const float* b2 = (const float*)d_in[7];
  const float* W3 = (const float*)d_in[8];
  float* out = (float*)d_out;

  // Workspace layout (bf16 weight copies, ~2.2 MB total; L2-resident on MI455X)
  char* ws = (char*)d_ws;
  unsigned short* W0f = (unsigned short*)(ws + 0);        //  512 x 64
  unsigned short* W0r = (unsigned short*)(ws + 65536);    //   64 x 512
  unsigned short* W1f = (unsigned short*)(ws + 131072);   //  512 x 512
  unsigned short* W1r = (unsigned short*)(ws + 655360);
  unsigned short* W2f = (unsigned short*)(ws + 1179648);
  unsigned short* W2r = (unsigned short*)(ws + 1703936);

  prep_weights<<<(DIN * HID + 255) / 256, 256, 0, stream>>>(W0, W0f, W0r, DIN, HID);
  prep_weights<<<(HID * HID + 255) / 256, 256, 0, stream>>>(W1, W1f, W1r, HID, HID);
  prep_weights<<<(HID * HID + 255) / 256, 256, 0, stream>>>(W2, W2f, W2r, HID, HID);

  size_t shmem = (size_t)WPB * 3 * TILE_ELEMS * sizeof(unsigned short);  // 192 KB
  (void)hipFuncSetAttribute((const void*)hnn_fused,
                            hipFuncAttributeMaxDynamicSharedMemorySize, (int)shmem);
  int blocks = BATCH / (16 * WPB);
  hnn_fused<<<blocks, WPB * 32, shmem, stream>>>(x, b0, b1, b2, W3,
                                                 W0f, W0r, W1f, W1r, W2f, W2r, out);
}
